// ConLoss_36928128811312
// MI455X (gfx1250) — matrix operand order repaired and checked by reading, not compiled, not run on hardware
//
#include <hip/hip_runtime.h>

// Problem constants (fixed by the reference).
#define B_ 8
#define N_ 128
#define T_ 8
#define L_ 6
#define D_ 256
#define P_ (N_*T_)          // 1024

#define INV_T 14.285714285714286f   // 1/0.07
#define MCAP  14.285714285714286f   // fixed log-sum-exp cap (|z| <= 1/TEMP)

typedef __attribute__((ext_vector_type(16))) __bf16 v16bf;
typedef __attribute__((ext_vector_type(8)))  float  v8f;

// ---------------------------------------------------------------- init
__global__ void kinit(float* lsum, int* Kb, int* ngt) {
  int t = threadIdx.x;
  if (t < B_*L_) lsum[t] = 0.f;
  if (t < B_)    { Kb[t] = 0; ngt[t] = 0; }
}

// ------------------------------------------------- per-row label metadata
__global__ __launch_bounds__(256) void prep_rows(
    const int* __restrict__ thing, const int* __restrict__ label,
    int* __restrict__ labx, int* __restrict__ posc, int* __restrict__ negc,
    int* __restrict__ lastc, int* __restrict__ Kb, int* __restrict__ ngt) {
  int gid = blockIdx.x*256 + threadIdx.x;
  if (gid >= B_*P_) return;
  int b = gid / P_;
  int p = gid & (P_-1);
  int v = (thing[gid] != 0);
  int lab = label[gid];
  labx[gid] = v ? lab : -1;
  if (v) atomicAdd(&Kb[b], 1);
  if ((p & (T_-1)) == 0) {                 // gate: rows (b,n) with any gt t
    int any = 0;
    #pragma unroll
    for (int t = 0; t < T_; ++t) any |= (thing[gid+t] != 0);
    if (any) atomicAdd(&ngt[b], 1);
  }
  const int* th = thing + b*P_;
  const int* lb = label + b*P_;
  int pc = 0, nc = 0, last = 0;
  for (int q = 0; q < P_; ++q) {
    int vq   = (th[q] != 0);
    int same = (lb[q] == lab);
    if (v && vq) {
      if (same) pc += (q != p);
      else      { ++nc; last = q; }
    }
  }
  posc[gid] = pc; negc[gid] = nc; lastc[gid] = last;   // last pre-clamped >= 0
}

// -------------------------------------------- normalize rows, cast to bf16
// Output layout: wn[(b*L + l)*P + p][d]  (row-major, GEMM friendly).
__global__ __launch_bounds__(256) void normalize_k(
    const float* __restrict__ meta, __bf16* __restrict__ wn) {
  int gid  = blockIdx.x*256 + threadIdx.x;
  int lane = gid & 31;
  int row  = gid >> 5;                     // one wave per row
  if (row >= B_*L_*P_) return;
  int p  = row & (P_-1);
  int bl = row / P_;
  int l  = bl % L_;
  int b  = bl / L_;
  const float* src = meta + (((size_t)(b*P_ + p)*L_ + l)*D_) + lane*8;
  float x[8];
  *(float4*)(x)     = *(const float4*)(src);
  *(float4*)(x + 4) = *(const float4*)(src + 4);
  float ss = 0.f;
  #pragma unroll
  for (int i = 0; i < 8; ++i) ss += x[i]*x[i];
  #pragma unroll
  for (int m = 16; m >= 1; m >>= 1) ss += __shfl_xor(ss, m, 32);
  float sc = 1.f / fmaxf(sqrtf(ss), 1e-12f);
  union { __bf16 h[8]; uint4 q; } u;
  #pragma unroll
  for (int i = 0; i < 8; ++i) u.h[i] = (__bf16)(x[i]*sc);
  *(uint4*)(wn + (size_t)row*D_ + lane*8) = u.q;
}

// -------------------------------------- deterministic random positive pick
__global__ __launch_bounds__(256) void sel_k(
    const int* __restrict__ labx, const int* __restrict__ posc,
    int* __restrict__ selc) {
  int gid = blockIdx.x*256 + threadIdx.x;
  if (gid >= B_*L_*P_) return;
  int p  = gid & (P_-1);
  int bl = gid / P_;
  int b  = bl / L_;
  int lab = labx[b*P_ + p];
  int pc  = posc[b*P_ + p];
  int sel = p;                              // diag fallback when no positives
  if (lab >= 0 && pc > 0) {
    unsigned h = (unsigned)gid * 2654435761u;
    h ^= h >> 16; h *= 2246822519u; h ^= h >> 13; h *= 3266489917u; h ^= h >> 16;
    float u01 = (float)(h >> 8) * (1.0f/16777216.0f);
    int r = (int)(u01 * (float)pc);
    r = min(r, pc - 1);
    const int* lx = labx + b*P_;
    int cnt = 0;
    for (int q = 0; q < P_; ++q) {
      if (lx[q] == lab && q != p) {
        if (cnt == r) { sel = q; break; }
        ++cnt;
      }
    }
  }
  selc[gid] = sel;
}

// ---------------- fused bf16-WMMA Gram matrix + masked log-sum-exp loss ---
#define CPAN 64        // columns per LDS panel
#define LDSS 264       // padded row stride (bf16 elems) -> conflict-free B reads
#define PANEL_ELEMS (CPAN*LDSS)

// Issue one panel (64 rows x 512 B) of async memory->LDS copies.
// 8 GLOBAL_LOAD_ASYNC_TO_LDS_B128 per thread; ASYNCcnt-tracked, no VGPR staging.
// NOTE: the LDS byte address is derived from the REAL shared-memory pointer
// (ptrtoint) and the pointer is also passed as an operand, so the allocation
// escapes into the asm and the "memory" clobber is honored by alias analysis.
__device__ __forceinline__ void fill_panel_async(
    const __bf16* __restrict__ wnBL, int c0, __bf16* smBuf, int tid) {
  #pragma unroll
  for (int it = 0; it < 8; ++it) {
    int c   = tid + it*256;                 // chunk id in [0, 2048)
    int rr  = c >> 5;                       // panel row
    int seg = c & 31;                       // 16-byte segment within row
    __bf16* dst = smBuf + rr*LDSS + seg*8;
    unsigned lds_addr = (unsigned)(uintptr_t)dst;   // addr[31:0] = LDS offset
    const __bf16* g = wnBL + (size_t)(c0 + rr)*D_ + seg*8;
    asm volatile("global_load_async_to_lds_b128 %0, %1, off"
                 :: "v"(lds_addr), "v"(g), "v"(dst) : "memory");
  }
}

__global__ __launch_bounds__(256) void gram_loss(
    const __bf16* __restrict__ wn,  const int* __restrict__ labx,
    const int* __restrict__ negc,   const int* __restrict__ lastc,
    const int* __restrict__ selc,   const int* __restrict__ Kb,
    float* __restrict__ lsum) {
  __shared__ __bf16 sm[2*PANEL_ELEMS];      // double-buffered: 67,584 bytes
  const int bl    = blockIdx.y;
  const int b     = bl / L_;
  const int tid   = threadIdx.x;
  const int wave  = tid >> 5;
  const int lane  = tid & 31;
  const int l16   = lane & 15;
  const int khalf = lane >> 4;
  const int rowBase = blockIdx.x*128 + wave*16;
  const __bf16* wnBL = wn + (size_t)bl*P_*D_;

  // Kick off panel 0 before touching anything else.
  fill_panel_async(wnBL, 0, sm, tid);

  // A fragments (rows rowBase..+15, K=0..255) held in registers all kernel.
  // ISA layout: lane m holds row m; lane-half selects K-octet within 32-chunk.
  v16bf A[8];
  {
    const __bf16* arow = wnBL + (size_t)(rowBase + l16)*D_;
    #pragma unroll
    for (int f = 0; f < 8; ++f) {
      union { v16bf v; uint4 q[2]; } ua;
      ua.q[0] = *(const uint4*)(arow + 32*f +      khalf*8);
      ua.q[1] = *(const uint4*)(arow + 32*f + 16 + khalf*8);
      A[f] = ua.v;
    }
  }
  // Row metadata: C/D layout puts row = rowBase + vi + 8*(lane>=16).
  int labr[8], selr[8], lastr[8];
  #pragma unroll
  for (int vi = 0; vi < 8; ++vi) {
    int r = rowBase + vi + khalf*8;
    labr[vi]  = labx[b*P_ + r];
    selr[vi]  = selc[bl*P_ + r];
    lastr[vi] = lastc[b*P_ + r];
  }
  float S[8], ZP[8], ZL[8];
  #pragma unroll
  for (int vi = 0; vi < 8; ++vi) { S[vi] = 0.f; ZP[vi] = 0.f; ZL[vi] = 0.f; }

  for (int panel = 0; panel < P_/CPAN; ++panel) {
    const int c0 = panel*CPAN;
    __bf16* smCur = sm + ((panel & 1) ? PANEL_ELEMS : 0);
    __bf16* smNxt = sm + ((panel & 1) ? 0 : PANEL_ELEMS);
    const bool havePrefetch = (panel + 1 < P_/CPAN);
    if (havePrefetch)                        // prefetch next panel, other buffer
      fill_panel_async(wnBL, c0 + CPAN, smNxt, tid);
    // Async loads complete in order: <=8 outstanding => current panel landed.
    if (havePrefetch) asm volatile("s_wait_asynccnt 0x8" ::: "memory");
    else              asm volatile("s_wait_asynccnt 0x0" ::: "memory");
    __syncthreads();                         // all waves' fills visible

    #pragma unroll
    for (int st = 0; st < CPAN/16; ++st) {
      int col  = c0 + st*16 + l16;           // this lane's column in the tile
      int labc = labx[b*P_ + col];
      v8f acc = {0.f,0.f,0.f,0.f,0.f,0.f,0.f,0.f};
      #pragma unroll
      for (int f = 0; f < 8; ++f) {          // K = 256 in 8 x K32 WMMAs
        union { v16bf v; uint4 q[2]; } ub;
        const __bf16* bp = smCur + (st*16 + l16)*LDSS + 32*f + khalf*8;
        ub.q[0] = *(const uint4*)(bp);
        ub.q[1] = *(const uint4*)(bp + 16);
        acc = __builtin_amdgcn_wmma_f32_16x16x32_bf16(
                  false, A[f], false, ub.v, (short)0, acc, false, false);
      }
      #pragma unroll
      for (int vi = 0; vi < 8; ++vi) {       // fully branchless accumulation
        float z = acc[vi] * INV_T;
        float e = __expf(z - MCAP);          // unconditional: no EXEC juggling
        bool  neg = (labc >= 0) & (labr[vi] >= 0) & (labc != labr[vi]);
        S[vi]  = fmaf(neg ? 1.f : 0.f, e, S[vi]);
        ZP[vi] += (col == selr[vi])  ? z : 0.f;
        ZL[vi] += (col == lastr[vi]) ? z : 0.f;
      }
    }
    __syncthreads();                         // everyone done reading this buffer
  }
  // Reduce each row across its 16-lane half.
  #pragma unroll
  for (int vi = 0; vi < 8; ++vi) {
    #pragma unroll
    for (int m = 8; m >= 1; m >>= 1) {
      S[vi]  += __shfl_xor(S[vi],  m, 16);
      ZP[vi] += __shfl_xor(ZP[vi], m, 16);
      ZL[vi] += __shfl_xor(ZL[vi], m, 16);
    }
  }
  if (l16 == 0) {
    float local = 0.f;
    #pragma unroll
    for (int vi = 0; vi < 8; ++vi) {
      if (labr[vi] >= 0) {                   // only valid rows contribute
        int   r   = rowBase + vi + khalf*8;
        int   nc  = negc[b*P_ + r];
        int   Kv  = Kb[b];
        float pad = fmaxf((float)(Kv - 1 - nc), 0.f);
        float pt  = (nc > 0) ? pad*__expf(ZL[vi] - MCAP) : 0.f;
        local += MCAP + __logf(__expf(ZP[vi] - MCAP) + S[vi] + pt) - ZP[vi];
      }
    }
    atomicAdd(&lsum[bl], local);
  }
}

// ---------------------------------------------------------------- finalize
__global__ void final_k(const float* __restrict__ lsum, const int* __restrict__ Kb,
                        const int* __restrict__ ngt, float* __restrict__ out) {
  if (threadIdx.x == 0 && blockIdx.x == 0) {
    float s = 0.f;
    for (int t = 0; t < B_*L_; ++t) {
      int b = t / L_;
      if (ngt[b] >= 2) s += lsum[t] / (float)max(Kb[b], 1);
    }
    out[0] = s;
  }
}

// ---------------------------------------------------------------- launcher
extern "C" void kernel_launch(void* const* d_in, const int* in_sizes, int n_in,
                              void* d_out, int out_size, void* d_ws, size_t ws_size,
                              hipStream_t stream) {
  (void)in_sizes; (void)n_in; (void)out_size; (void)ws_size;
  const float* meta  = (const float*)d_in[0];
  const int*   thing = (const int*)d_in[1];
  const int*   label = (const int*)d_in[2];
  float* out = (float*)d_out;

  char* w = (char*)d_ws;
  size_t o = 0;
  __bf16* wn   = (__bf16*)(w + o); o += (size_t)B_*L_*P_*D_*sizeof(__bf16); // ~24 MB
  int*   labx  = (int*)(w + o);    o += (size_t)B_*P_*4;
  int*   posc  = (int*)(w + o);    o += (size_t)B_*P_*4;
  int*   negcv = (int*)(w + o);    o += (size_t)B_*P_*4;
  int*   lastv = (int*)(w + o);    o += (size_t)B_*P_*4;
  int*   selc  = (int*)(w + o);    o += (size_t)B_*L_*P_*4;
  int*   Kb    = (int*)(w + o);    o += (size_t)B_*4;
  int*   ngt   = (int*)(w + o);    o += (size_t)B_*4;
  float* lsum  = (float*)(w + o);  o += (size_t)B_*L_*4;

  hipLaunchKernelGGL(kinit, dim3(1), dim3(64), 0, stream, lsum, Kb, ngt);
  hipLaunchKernelGGL(prep_rows, dim3((B_*P_ + 255)/256), dim3(256), 0, stream,
                     thing, label, labx, posc, negcv, lastv, Kb, ngt);
  hipLaunchKernelGGL(normalize_k, dim3((B_*L_*P_*32 + 255)/256), dim3(256), 0, stream,
                     meta, wn);
  hipLaunchKernelGGL(sel_k, dim3((B_*L_*P_ + 255)/256), dim3(256), 0, stream,
                     labx, posc, selc);
  hipLaunchKernelGGL(gram_loss, dim3(P_/128, B_*L_), dim3(256), 0, stream,
                     wn, labx, negcv, lastv, selc, Kb, lsum);
  hipLaunchKernelGGL(final_k, dim3(1), dim3(32), 0, stream, lsum, Kb, ngt, out);
}
